// SRUDecoder_Tiling_87754771792271
// MI455X (gfx1250) — matrix operand order repaired and checked by reading, not compiled, not run on hardware
//
#include <hip/hip_runtime.h>

// ---------------------------------------------------------------------------
// Problem constants (from reference): T=1024, B=32, DIN=256, H=256, AA=21, L=2
// ---------------------------------------------------------------------------
#define T_DIM 1024
#define B_DIM 32
#define DIN_DIM 256
#define H_DIM 256
#define AA_DIM 21
#define ROWS (B_DIM * T_DIM)      // 32768 rows (b*T + t ordering)

typedef __attribute__((ext_vector_type(16))) __bf16         bf16x16;
typedef __attribute__((ext_vector_type(8)))  float          f32x8;
typedef __attribute__((ext_vector_type(8)))  unsigned short u16x8;
typedef __attribute__((ext_vector_type(16))) unsigned short u16x16;
typedef __attribute__((ext_vector_type(4)))  unsigned int   u32x4;
typedef __attribute__((ext_vector_type(8)))  unsigned int   u32x8;

__device__ __forceinline__ unsigned short f32_to_bf16_rne(float f) {
    unsigned int u = __float_as_uint(f);
    u += 0x7FFFu + ((u >> 16) & 1u);          // round-to-nearest-even
    return (unsigned short)(u >> 16);
}

__device__ __forceinline__ bf16x16 frag_from(u16x8 lo, u16x8 hi) {
    u16x16 v = __builtin_shufflevector(lo, hi, 0,1,2,3,4,5,6,7,8,9,10,11,12,13,14,15);
    return __builtin_bit_cast(bf16x16, v);
}

// ---------------------------------------------------------------------------
// TDM: DMA a 2D tile (rows x Khalves of bf16) from global into LDS with
// descriptor-driven row padding (pad_amount DWORDs appended per row).
// D# packing per CDNA5 ISA ch.8 (group0 128b, group1 256b; groups 2/3 NULL
// for 2D tensors -> 2-operand tensor_load_to_lds form).
// ---------------------------------------------------------------------------
__device__ __forceinline__ void tdm_load_2d_to_lds(const void* gsrc, unsigned lds_byte_addr,
                                                   int Khalves, int rows,
                                                   int pad_interval_code, int pad_amount_code)
{
    unsigned long long ga = (unsigned long long)gsrc;
    u32x4 g0;
    g0[0] = 1u;                                             // count=1, user load desc
    g0[1] = lds_byte_addr;                                  // lds_addr [63:32]
    g0[2] = (unsigned)(ga & 0xFFFFFFFFu);                   // global_addr [95:64]
    g0[3] = (unsigned)((ga >> 32) & 0x1FFFFFFu) | (2u << 30); // global_addr hi | type=2

    u32x8 g1;
    g1[0] = (1u << 16)                                      // data_size = 1 (2 bytes)
          | (1u << 20)                                      // pad_enable
          | ((unsigned)pad_interval_code << 22)
          | ((unsigned)pad_amount_code << 25);
    g1[1] = ((unsigned)Khalves & 0xFFFFu) << 16;            // tensor_dim0 lo
    g1[2] = ((unsigned)rows & 0xFFFFu) << 16;               // tensor_dim0 hi=0 | tensor_dim1 lo
    g1[3] = ((unsigned)Khalves & 0xFFFFu) << 16;            // tensor_dim1 hi=0 | tile_dim0
    g1[4] = (unsigned)rows & 0xFFFFu;                       // tile_dim1 | tile_dim2=0
    g1[5] = (unsigned)Khalves;                              // tensor_dim0_stride lo
    g1[6] = 0u;                                             // stride hi | dim1_stride lo
    g1[7] = 0u;

    asm volatile("tensor_load_to_lds %0, %1" :: "s"(g0), "s"(g1) : "memory");
}

// ---------------------------------------------------------------------------
// f32 -> bf16 elementwise convert
// ---------------------------------------------------------------------------
__global__ void k_cvt(const float* __restrict__ in, unsigned short* __restrict__ out, long n) {
    long i = (long)blockIdx.x * blockDim.x + threadIdx.x;
    if (i < n) out[i] = f32_to_bf16_rne(in[i]);
}

// f32 (K x N) row-major  ->  bf16 (N x K) row-major (transpose + convert)
__global__ void k_cvt_t(const float* __restrict__ in, unsigned short* __restrict__ out, int K, int N) {
    long i = (long)blockIdx.x * blockDim.x + threadIdx.x;
    if (i >= (long)K * N) return;
    int k = (int)(i / N);
    int n = (int)(i % N);
    out[(long)n * K + k] = f32_to_bf16_rne(in[i]);
}

// ---------------------------------------------------------------------------
// bf16 GEMM, f32 accumulate:  C[M x N] = A[M x K] * BT[N x K]^T (+ bias)
// Block = 256 threads (8 waves) covering a 256(M) x 64(N) macro-tile.
// The block's 64 x K B slab is loop-invariant: staged ONCE into LDS by a
// single tensor_load_to_lds DMA (row pitch K+8 halves -> ds_load_b128 reads
// are bank-conflict free), then each wave computes a 32x64 tile:
// 2 A fragments (global b128) x 4 B fragments (LDS b128) = 8 WMMAs / k-step.
//   A frag (16x32): lane L = row M, group g: halves j<8 -> K=kk+g*8+j,
//                   j>=8 -> K=kk+16+g*8+(j-8)
//   B frag (32x16): lane L = col N, group g: same K mapping as A frag
//   C/D (16x16 f32): VGPR r, lane L -> M = 8*g + r, N = L%16
// ---------------------------------------------------------------------------
__global__ void __launch_bounds__(256)
k_gemm_bf16(const unsigned short* __restrict__ A,   // M x K bf16
            const unsigned short* __restrict__ BT,  // N x K bf16
            const float* __restrict__ bias,         // N or null
            float* __restrict__ C,                  // M x N f32
            int M, int N, int K)
{
    __shared__ unsigned short sB[64 * (512 + 8)];   // max K=512, pitch K+8 halves

    const int pitch  = K + 8;
    const int tilesN = N >> 6;
    const int bm     = blockIdx.x / tilesN;         // 256-row macro-block
    const int tn     = blockIdx.x % tilesN;         // 64-col group
    const int lane   = threadIdx.x & 31;
    const int w      = threadIdx.x >> 5;            // wave 0..7
    const int lan16  = lane & 15;
    const int g      = lane >> 4;

    if (w == 0) {
        // one DMA for the whole block's B slab; row bytes = K*2 = 8<<ic
        const unsigned ldsAddr = (unsigned)(unsigned long long)(const void*)sB;
        const int ic = (K == 512) ? 7 : 6;          // pad_interval: 1024B / 512B rows
        tdm_load_2d_to_lds(BT + (long)tn * 64 * K, ldsAddr, K, 64, ic, /*pad=4 dwords*/3);
        __builtin_amdgcn_s_wait_tensorcnt(0);
    }
    __syncthreads();

    const int m0 = bm * 256 + w * 32;
    const unsigned short* arow0 = A + (long)(m0 +  0 + lan16) * K + g * 8;
    const unsigned short* arow1 = A + (long)(m0 + 16 + lan16) * K + g * 8;

    f32x8 acc[2][4] = {};

    for (int kk = 0; kk < K; kk += 32) {
        bf16x16 a0 = frag_from(*(const u16x8*)(arow0 + kk), *(const u16x8*)(arow0 + kk + 16));
        bf16x16 a1 = frag_from(*(const u16x8*)(arow1 + kk), *(const u16x8*)(arow1 + kk + 16));
#pragma unroll
        for (int t = 0; t < 4; ++t) {
            const int off = (t * 16 + lan16) * pitch + kk + g * 8;
            bf16x16 b = frag_from(*(const u16x8*)&sB[off], *(const u16x8*)&sB[off + 16]);
            acc[0][t] = __builtin_amdgcn_wmma_f32_16x16x32_bf16(false, a0, false, b, (short)0, acc[0][t], false, false);
            acc[1][t] = __builtin_amdgcn_wmma_f32_16x16x32_bf16(false, a1, false, b, (short)0, acc[1][t], false, false);
        }
    }

#pragma unroll
    for (int s = 0; s < 2; ++s) {
#pragma unroll
        for (int t = 0; t < 4; ++t) {
            const int col = tn * 64 + t * 16 + lan16;
            const float bv = bias ? bias[col] : 0.0f;
#pragma unroll
            for (int r = 0; r < 8; ++r) {
                const int row = m0 + s * 16 + g * 8 + r;
                C[(long)row * N + col] = acc[s][t][r] + bv;
            }
        }
    }
}

// ---------------------------------------------------------------------------
// SRU recurrence, one thread per (dir, b, h).  Reads U (rows x 2048 f32,
// row = b*T+t, col = dir*4H + chunk*H + h), writes bf16 h-out (rows x 2H).
// chunks: 0=candidate 1=forget 2=reset 3=highway; r uses the OLD cell state.
// ---------------------------------------------------------------------------
__global__ void k_sru_scan(const float* __restrict__ U,
                           const float* __restrict__ wc,      // (2,2,H)
                           const float* __restrict__ bias,    // (2,2,H)
                           unsigned short* __restrict__ Hout) // rows x 2H bf16
{
    int idx = blockIdx.x * blockDim.x + threadIdx.x;      // over 2*B*H
    if (idx >= 2 * B_DIM * H_DIM) return;
    const int h   = idx % H_DIM;
    const int b   = (idx / H_DIM) % B_DIM;
    const int dir = idx / (H_DIM * B_DIM);

    const float vf = wc  [(dir * 2 + 0) * H_DIM + h];
    const float vr = wc  [(dir * 2 + 1) * H_DIM + h];
    const float bf = bias[(dir * 2 + 0) * H_DIM + h];
    const float br = bias[(dir * 2 + 1) * H_DIM + h];

    const long rowstride = 8 * H_DIM;                     // 2048 cols
    const float* Ub = U + (long)b * T_DIM * rowstride + (long)dir * 4 * H_DIM + h;
    unsigned short* Ob = Hout + ((long)b * T_DIM) * (2 * H_DIM) + dir * H_DIM + h;

    float c = 0.0f;
    for (int s = 0; s < T_DIM; ++s) {
        const int t = dir ? (T_DIM - 1 - s) : s;
        const float* u = Ub + (long)t * rowstride;
        const float u0 = u[0 * H_DIM];
        const float u1 = u[1 * H_DIM];
        const float u2 = u[2 * H_DIM];
        const float u3 = u[3 * H_DIM];
        const float f  = 1.0f / (1.0f + __expf(-(u1 + vf * c + bf)));
        const float c2 = f * c + (1.0f - f) * u0;
        const float r  = 1.0f / (1.0f + __expf(-(u2 + vr * c + br)));
        const float hv = r * c2 + (1.0f - r) * u3;
        c = c2;
        Ob[(long)t * (2 * H_DIM)] = f32_to_bf16_rne(hv);
    }
}

// ---------------------------------------------------------------------------
// Final projection to AA=21 classes + log_softmax.  One wave per row:
// lanes 0..20 each compute one logit (f32 dot over H), then wave-wide
// max / sum-exp reductions via shfl_xor over the 32-lane wave.
// ---------------------------------------------------------------------------
__global__ void k_logits_lsm(const float* __restrict__ X,   // rows x H
                             const float* __restrict__ W,   // AA x H
                             const float* __restrict__ bias,// AA
                             float* __restrict__ out,       // rows x AA
                             int rows)
{
    const int lane = threadIdx.x & 31;
    const int row  = blockIdx.x * (blockDim.x >> 5) + (threadIdx.x >> 5);
    if (row >= rows) return;

    const float* x = X + (long)row * H_DIM;
    float acc = 0.0f;
    if (lane < AA_DIM) {
        const float* w = W + (long)lane * H_DIM;
#pragma unroll 8
        for (int j = 0; j < H_DIM; ++j) acc = fmaf(x[j], w[j], acc);
        acc += bias[lane];
    }
    const float logit = (lane < AA_DIM) ? acc : -1e30f;

    float m = logit;
    for (int off = 16; off > 0; off >>= 1) m = fmaxf(m, __shfl_xor(m, off, 32));
    float e = (lane < AA_DIM) ? __expf(logit - m) : 0.0f;
    float s = e;
    for (int off = 16; off > 0; off >>= 1) s += __shfl_xor(s, off, 32);

    if (lane < AA_DIM) out[(long)row * AA_DIM + lane] = logit - m - __logf(s);
}

// ---------------------------------------------------------------------------
// Orchestration.  Workspace layout (all reused across stages):
//   U   : ROWS x 2048 f32  (U0 -> U1 -> fc1 output)      256 MB
//   Abf : ROWS x  512 bf16 (x_bf -> h0_bf -> h1_bf)       32 MB
//   Wbf : 2048 x 512 bf16  (W0^T -> W1^T -> fc1_w)         2 MB
// ---------------------------------------------------------------------------
extern "C" void kernel_launch(void* const* d_in, const int* in_sizes, int n_in,
                              void* d_out, int out_size, void* d_ws, size_t ws_size,
                              hipStream_t stream)
{
    (void)in_sizes; (void)n_in; (void)out_size; (void)ws_size;
    const float* x     = (const float*)d_in[0];
    // d_in[1] = hidden (unused, matches reference)
    const float* w_l0  = (const float*)d_in[2];
    const float* wc_l0 = (const float*)d_in[3];
    const float* b_l0  = (const float*)d_in[4];
    const float* w_l1  = (const float*)d_in[5];
    const float* wc_l1 = (const float*)d_in[6];
    const float* b_l1  = (const float*)d_in[7];
    const float* fc1_w = (const float*)d_in[8];
    const float* fc1_b = (const float*)d_in[9];
    const float* lp_w  = (const float*)d_in[10];
    const float* lp_b  = (const float*)d_in[11];
    float* out = (float*)d_out;

    char* ws = (char*)d_ws;
    float*          U   = (float*)ws;                                           // ROWS*2048 f32
    unsigned short* Abf = (unsigned short*)(ws + (size_t)ROWS * 2048 * 4);      // ROWS*512 bf16
    unsigned short* Wbf = (unsigned short*)(ws + (size_t)ROWS * 2048 * 4
                                               + (size_t)ROWS * 512 * 2);       // 2048*512 bf16

    auto nb = [](long n, int t) { return (unsigned)((n + t - 1) / t); };

    // ---- layer 0 ----
    k_cvt  <<<nb((long)ROWS * DIN_DIM, 256), 256, 0, stream>>>(x, Abf, (long)ROWS * DIN_DIM);
    k_cvt_t<<<nb((long)DIN_DIM * 2048, 256), 256, 0, stream>>>(w_l0, Wbf, DIN_DIM, 2048);
    {
        const int M = ROWS, N = 2048, K = DIN_DIM;
        k_gemm_bf16<<<(M / 256) * (N / 64), 256, 0, stream>>>(Abf, Wbf, nullptr, U, M, N, K);
    }
    k_sru_scan<<<nb(2 * B_DIM * H_DIM, 256), 256, 0, stream>>>(U, wc_l0, b_l0, Abf);

    // ---- layer 1 ----
    k_cvt_t<<<nb((long)512 * 2048, 256), 256, 0, stream>>>(w_l1, Wbf, 512, 2048);
    {
        const int M = ROWS, N = 2048, K = 512;
        k_gemm_bf16<<<(M / 256) * (N / 64), 256, 0, stream>>>(Abf, Wbf, nullptr, U, M, N, K);
    }
    k_sru_scan<<<nb(2 * B_DIM * H_DIM, 256), 256, 0, stream>>>(U, wc_l1, b_l1, Abf);

    // ---- fc1: h1 @ fc1_w.T ; fc1_w (H x 2H) row-major is already N x K ----
    k_cvt<<<nb((long)H_DIM * 512, 256), 256, 0, stream>>>(fc1_w, Wbf, (long)H_DIM * 512);
    {
        const int M = ROWS, N = H_DIM, K = 512;
        k_gemm_bf16<<<(M / 256) * (N / 64), 256, 0, stream>>>(Abf, Wbf, fc1_b, U, M, N, K);
    }

    // ---- logits + log_softmax ----
    k_logits_lsm<<<nb((long)ROWS * 32, 256), 256, 0, stream>>>(U, lp_w, lp_b, out, ROWS);
}
